// VectorQuantizer1D_8598524526949
// MI455X (gfx1250) — matrix operand order repaired and checked by reading, not compiled, not run on hardware
//
#include <hip/hip_runtime.h>

// ---------------------------------------------------------------------------
// VQ codebook search for MI455X (gfx1250, wave32, WMMA bf16 16x16x32).
// B=16 L=2048 D=512 K=8192  ->  N=32768 tokens.
// Score: cnorm[k] - 2 * (x . c_k)   (||x||^2 dropped: constant per row)
// Main kernel: 2x2 register-blocked WMMA (32 rows x 32 codes per wave)
//              -> 2 b128 loads per WMMA instead of 4.
// ---------------------------------------------------------------------------

typedef __attribute__((ext_vector_type(16))) __bf16 v16bf;
typedef __attribute__((ext_vector_type(4)))  __bf16 v4bf;
typedef __attribute__((ext_vector_type(2)))  __bf16 v2bf;
typedef __attribute__((ext_vector_type(8)))  float  v8f;

#define DDIM 512
#define KCODES 8192
#define NTOK 32768
#define MT 64          // tokens per block in main kernel
#define LDSROW 520     // bf16 elems per LDS row (512 + 8 pad -> 1040B, rotates banks)

// ---------------------------------------------------------------------------
// Kernel 1: codebook fp32 -> bf16 + per-code squared norm. One block per code.
// ---------------------------------------------------------------------------
__global__ __launch_bounds__(256) void vq_prep_cb(const float* __restrict__ cb,
                                                  __bf16* __restrict__ cbh,
                                                  float* __restrict__ cnorm) {
    const int c = blockIdx.x;
    const int t = threadIdx.x;
    const float2 v = ((const float2*)(cb + (size_t)c * DDIM))[t];
    v2bf p;
    p.x = (__bf16)v.x;
    p.y = (__bf16)v.y;
    ((v2bf*)(cbh + (size_t)c * DDIM))[t] = p;

    float s = v.x * v.x + v.y * v.y;
    #pragma unroll
    for (int m = 16; m >= 1; m >>= 1) s += __shfl_xor(s, m, 32);
    __shared__ float r[8];
    if ((t & 31) == 0) r[t >> 5] = s;
    __syncthreads();
    if (t == 0) {
        float tot = 0.f;
        #pragma unroll
        for (int i = 0; i < 8; ++i) tot += r[i];
        cnorm[c] = tot;
    }
}

// ---------------------------------------------------------------------------
// Kernel 2: x fp32 -> bf16, elementwise, float4 granularity.
// ---------------------------------------------------------------------------
__global__ __launch_bounds__(256) void vq_prep_x(const float* __restrict__ x,
                                                 __bf16* __restrict__ xh) {
    const size_t i = (size_t)blockIdx.x * 256 + threadIdx.x;
    const float4 v = ((const float4*)x)[i];
    v4bf p;
    p.x = (__bf16)v.x; p.y = (__bf16)v.y; p.z = (__bf16)v.z; p.w = (__bf16)v.w;
    ((v4bf*)xh)[i] = p;
}

// ---------------------------------------------------------------------------
// Kernel 3: main WMMA search. 512 blocks x 256 threads (8 wave32s).
//   - stage 64x512 bf16 x-tile into LDS once (64KB, padded rows)
//   - waves arranged 2(M: 32 rows) x 4(N: 32 codes); each wave holds a
//     2x2 accumulator grid -> 4 WMMAs per k-step off 2 A + 2 B fragments
//   - 64 iterations of 128-code tiles over the L2-resident bf16 codebook
//   - per-lane running argmin, shfl_xor half-wave reduce, 4-way LDS combine
// ---------------------------------------------------------------------------
__global__ __launch_bounds__(256) void vq_main(const __bf16* __restrict__ xh,
                                               const __bf16* __restrict__ cbh,
                                               const float* __restrict__ cnorm,
                                               int* __restrict__ idxout,
                                               float* __restrict__ idxf) {
    __shared__ __bf16 xs[MT * LDSROW];
    __shared__ float redD[4][MT];
    __shared__ int   redI[4][MT];

    const int tid = threadIdx.x;
    const int m0  = blockIdx.x * MT;

    // ---- stage x-tile: 4 threads per row, 8 x b128 each ----
    {
        const int r    = tid >> 2;
        const int cseg = (tid & 3) * 128;   // bf16 elements
        const uint4* src = (const uint4*)(xh + (size_t)(m0 + r) * DDIM + cseg);
        uint4* dst = (uint4*)(&xs[r * LDSROW + cseg]);
        #pragma unroll
        for (int i = 0; i < 8; ++i) dst[i] = src[i];
    }
    __syncthreads();

    const int wave = tid >> 5;
    const int lane = tid & 31;
    const int lr   = lane & 15;
    const int hi   = lane >> 4;          // lane half: 0 or 1
    const int wm   = wave & 1;           // M group: rows wm*32 .. +31
    const int wn   = wave >> 1;          // N group: codes wn*32 .. +31 (mod 128)

    const __bf16* arow0 = &xs[(wm * 32 + lr) * LDSROW];
    const __bf16* arow1 = &xs[(wm * 32 + 16 + lr) * LDSROW];

    // best slots: 0..7 -> rows wm*32 + v + hi*8 ; 8..15 -> +16
    float bestD[16];
    int   bestI[16];
    #pragma unroll
    for (int s = 0; s < 16; ++s) { bestD[s] = 3.4e38f; bestI[s] = 0; }

    for (int ct = 0; ct < KCODES / 128; ++ct) {
        const int code0 = ct * 128 + wn * 32 + lr;
        const int code1 = code0 + 16;
        const __bf16* brow0 = cbh + (size_t)code0 * DDIM;
        const __bf16* brow1 = cbh + (size_t)code1 * DDIM;
        const float cn0 = cnorm[code0];
        const float cn1 = cnorm[code1];

        v8f a00 = {0.f,0.f,0.f,0.f,0.f,0.f,0.f,0.f};
        v8f a01 = a00, a10 = a00, a11 = a00;

        #pragma unroll 2
        for (int ks = 0; ks < 16; ++ks) {
            // A fragments (16x32 bf16): lanes<16 hold K 0-7 & 16-23
            union { uint4 q[2]; v16bf v; } A0, A1, B0, B1;
            const int ao = ks * 32 + hi * 8;
            A0.q[0] = *(const uint4*)(arow0 + ao);
            A0.q[1] = *(const uint4*)(arow0 + ao + 16);
            A1.q[0] = *(const uint4*)(arow1 + ao);
            A1.q[1] = *(const uint4*)(arow1 + ao + 16);
            // B fragments (32x16 bf16): lane holds column lr, K = hi*8 .. +7 & 16+hi*8 .. +7
            const int bo = ks * 32 + hi * 8;
            const uint4* b0p = (const uint4*)(brow0 + bo);
            const uint4* b1p = (const uint4*)(brow1 + bo);
            B0.q[0] = b0p[0]; B0.q[1] = b0p[2];
            B1.q[0] = b1p[0]; B1.q[1] = b1p[2];

            a00 = __builtin_amdgcn_wmma_f32_16x16x32_bf16(
                false, A0.v, false, B0.v, (short)0, a00, false, false);
            a01 = __builtin_amdgcn_wmma_f32_16x16x32_bf16(
                false, A0.v, false, B1.v, (short)0, a01, false, false);
            a10 = __builtin_amdgcn_wmma_f32_16x16x32_bf16(
                false, A1.v, false, B0.v, (short)0, a10, false, false);
            a11 = __builtin_amdgcn_wmma_f32_16x16x32_bf16(
                false, A1.v, false, B1.v, (short)0, a11, false, false);
        }

        #pragma unroll
        for (int v = 0; v < 8; ++v) {
            // m-group 0 (rows wm*32 + v + hi*8)
            {
                float cd = fmaf(-2.0f, a00[v], cn0);
                int   ci = code0;
                const float d1 = fmaf(-2.0f, a01[v], cn1);
                if (d1 < cd) { cd = d1; ci = code1; }     // tie -> keep lower idx
                if (cd < bestD[v]) { bestD[v] = cd; bestI[v] = ci; }
            }
            // m-group 1 (rows wm*32 + 16 + v + hi*8)
            {
                float cd = fmaf(-2.0f, a10[v], cn0);
                int   ci = code0;
                const float d1 = fmaf(-2.0f, a11[v], cn1);
                if (d1 < cd) { cd = d1; ci = code1; }
                if (cd < bestD[8 + v]) { bestD[8 + v] = cd; bestI[8 + v] = ci; }
            }
        }
    }

    // ---- reduce over the 16 lanes of each half (masks 1..8 stay in-half) ----
    #pragma unroll
    for (int s = 0; s < 16; ++s) {
        float d = bestD[s];
        int   i = bestI[s];
        #pragma unroll
        for (int m = 1; m < 16; m <<= 1) {
            const float od = __shfl_xor(d, m, 32);
            const int   oi = __shfl_xor(i, m, 32);
            if (od < d || (od == d && oi < i)) { d = od; i = oi; }
        }
        if (lr == 0) {
            const int row = wm * 32 + (s >> 3) * 16 + (s & 7) + hi * 8;
            redD[wn][row] = d;
            redI[wn][row] = i;
        }
    }
    __syncthreads();

    // ---- combine the four code partitions, emit index ----
    if (tid < MT) {
        float bd = redD[0][tid];
        int   bi = redI[0][tid];
        #pragma unroll
        for (int p = 1; p < 4; ++p) {
            const float d = redD[p][tid];
            const int   i = redI[p][tid];
            if (d < bd || (d == bd && i < bi)) { bd = d; bi = i; }
        }
        idxout[m0 + tid] = bi;
        idxf[m0 + tid]   = (float)bi;   // tuple tail in d_out (float dtype)
    }
}

// ---------------------------------------------------------------------------
// Kernel 4: gather fp32 codebook rows -> z_quantized. One block per token.
// ---------------------------------------------------------------------------
__global__ __launch_bounds__(128) void vq_gather(const int* __restrict__ idx,
                                                 const float* __restrict__ cb,
                                                 float* __restrict__ out) {
    const int n = blockIdx.x;
    const int t = threadIdx.x;          // 128 threads, float4 each = 512 floats
    const int c = idx[n];
    const float4 v = ((const float4*)(cb + (size_t)c * DDIM))[t];
    ((float4*)(out + (size_t)n * DDIM))[t] = v;
}

// ---------------------------------------------------------------------------
// Launch
// ---------------------------------------------------------------------------
extern "C" void kernel_launch(void* const* d_in, const int* in_sizes, int n_in,
                              void* d_out, int out_size, void* d_ws, size_t ws_size,
                              hipStream_t stream) {
    const float* x  = (const float*)d_in[0];   // (16,2048,512) fp32
    const float* cb = (const float*)d_in[1];   // (8192,512) fp32
    float* out = (float*)d_out;                // z (N*D floats) ++ indices (N floats)

    char* ws = (char*)d_ws;
    __bf16* cbh   = (__bf16*)ws;                                   //  8 MB
    __bf16* xh    = (__bf16*)(ws + (size_t)KCODES * DDIM * 2);     // 32 MB
    float*  cnorm = (float*)(ws + (size_t)KCODES * DDIM * 2
                                 + (size_t)NTOK * DDIM * 2);       // 32 KB
    int*    idx   = (int*)((char*)cnorm + (size_t)KCODES * 4);     // 128 KB

    float* idxf = out + (size_t)NTOK * DDIM;   // index tail of d_out

    vq_prep_cb<<<KCODES, 256, 0, stream>>>(cb, cbh, cnorm);
    vq_prep_x<<<(NTOK * DDIM) / (256 * 4), 256, 0, stream>>>(x, xh);
    vq_main<<<NTOK / MT, 256, 0, stream>>>(xh, cbh, cnorm, idx, idxf);
    vq_gather<<<NTOK, 128, 0, stream>>>(idx, cb, out);
}
